// RCNNDetector_64733747085453
// MI455X (gfx1250) — matrix operand order repaired and checked by reading, not compiled, not run on hardware
//
#include <hip/hip_runtime.h>
#include <hip/hip_bf16.h>

typedef _Float16 h16;
typedef __attribute__((ext_vector_type(16))) _Float16 v16h;
typedef __attribute__((ext_vector_type(8)))  _Float16 v8h;
typedef __attribute__((ext_vector_type(8)))  float    v8f;

#define NPROP   600
#define MPAD    608      // 19 macro-tiles of 32
#define CCH     512
#define HF      25
#define WFW     38
#define DIN     25088    // 512*7*7
#define HID     1024
#define KCLS    21
#define LOGPAD  32
#define DELN    84
#define DELPAD  96
#define TOPK_N  300

// ---------------------------------------------------------------------------
// Zero-fill fp16 (pad rows of A)
// ---------------------------------------------------------------------------
__global__ void zero_h16_kernel(h16* __restrict__ p, long cnt) {
    long t = (long)blockIdx.x * blockDim.x + threadIdx.x;
    if (t < cnt) p[t] = (h16)0.0f;
}

// ---------------------------------------------------------------------------
// LDS-tiled convert+transpose: fp32 W (K x N) -> fp16 Bt (Npad x K).
// Coalesced reads and writes; +1 pad column avoids LDS bank conflicts.
// Grid: (K/32, Npad/32), block (32, 8).
// ---------------------------------------------------------------------------
__global__ __launch_bounds__(256)
void transpose_f16_kernel(const float* __restrict__ W, h16* __restrict__ Bt,
                          int K, int N, int Npad) {
    __shared__ float tile[32][33];
    const int kb = blockIdx.x * 32;
    const int nb = blockIdx.y * 32;
    const int tx = threadIdx.x;      // 0..31
    const int ty = threadIdx.y;      // 0..7
    #pragma unroll
    for (int i = 0; i < 32; i += 8) {
        int k = kb + ty + i;
        int n = nb + tx;
        tile[ty + i][tx] = (n < N) ? W[(size_t)k * N + n] : 0.0f;
    }
    __syncthreads();
    #pragma unroll
    for (int i = 0; i < 32; i += 8) {
        int n = nb + ty + i;
        int k = kb + tx;
        Bt[(size_t)n * K + k] = (h16)tile[tx][ty + i];
    }
}

// ---------------------------------------------------------------------------
// ROI align (OUT=7, SR=2) -> fp16 A matrix row n, col c*49 + oy*7 + ox
// ---------------------------------------------------------------------------
__global__ void roi_align_kernel(const float* __restrict__ fmap,
                                 const float* __restrict__ props,
                                 h16* __restrict__ A,
                                 const int* __restrict__ img_h_p,
                                 const int* __restrict__ img_w_p) {
    int t = blockIdx.x * blockDim.x + threadIdx.x;
    const int total = NPROP * CCH * 49;
    if (t >= total) return;
    int n   = t / (CCH * 49);
    int rem = t % (CCH * 49);
    int c   = rem / 49;
    int p   = rem % 49;
    int oy  = p / 7, ox = p % 7;

    float img_h = (float)(*img_h_p), img_w = (float)(*img_w_p);
    float sy = (float)HF / img_h, sx = (float)WFW / img_w;
    float x1 = props[n*4+0] * sx - 0.5f;
    float y1 = props[n*4+1] * sy - 0.5f;
    float x2 = props[n*4+2] * sx - 0.5f;
    float y2 = props[n*4+3] * sy - 0.5f;
    float bh = (y2 - y1) * (1.0f / 7.0f);
    float bw = (x2 - x1) * (1.0f / 7.0f);
    const float* f = fmap + (size_t)c * HF * WFW;

    float sum = 0.0f;
    #pragma unroll
    for (int syi = 0; syi < 2; ++syi) {
        #pragma unroll
        for (int sxi = 0; sxi < 2; ++sxi) {
            float ty = (float)oy + ((float)syi + 0.5f) * 0.5f;
            float tx = (float)ox + ((float)sxi + 0.5f) * 0.5f;
            float y = y1 + ty * bh;
            float x = x1 + tx * bw;
            bool valid = (y > -1.0f) && (y < (float)HF) &&
                         (x > -1.0f) && (x < (float)WFW);
            float yc = fminf(fmaxf(y, 0.0f), (float)(HF - 1));
            float xc = fminf(fmaxf(x, 0.0f), (float)(WFW - 1));
            float y0f = floorf(yc), x0f = floorf(xc);
            float ly = yc - y0f, lx = xc - x0f;
            int y0 = (int)y0f, x0 = (int)x0f;
            int y1i = min(y0 + 1, HF - 1);
            int x1i = min(x0 + 1, WFW - 1);
            float v = (1.0f-ly)*(1.0f-lx)*f[y0 *WFW + x0 ]
                    + (1.0f-ly)*       lx *f[y0 *WFW + x1i]
                    +        ly *(1.0f-lx)*f[y1i*WFW + x0 ]
                    +        ly *       lx *f[y1i*WFW + x1i];
            sum += valid ? v : 0.0f;
        }
    }
    A[(size_t)n * DIN + (size_t)c * 49 + p] = (h16)(sum * 0.25f);
}

// ---------------------------------------------------------------------------
// WMMA fragment helpers (ISA 16-bit layouts)
// ---------------------------------------------------------------------------
__device__ __forceinline__ v16h load_afrag(const h16* p) {
    v8h a0 = *(const v8h*)p;
    v8h a1 = *(const v8h*)(p + 16);
    v16h r;
    #pragma unroll
    for (int i = 0; i < 8; ++i) { r[i] = a0[i]; r[i + 8] = a1[i]; }
    return r;
}
__device__ __forceinline__ v8f wmma16(v16h a, v16h b, v8f c) {
    return __builtin_amdgcn_wmma_f32_16x16x32_f16(false, a, false, b,
                                                  (short)0, c, false, false);
}

// ---------------------------------------------------------------------------
// WMMA GEMM: C(MPAD x Npad) = A(MPAD x K, f16) * Bt(Npad x K, f16)^T + bias
// One wave per 32x32 macro-tile (2x2 WMMA tiles): 4 WMMAs per 8 b128 loads,
// software double-buffered so next-K loads overlap current-K WMMAs.
// ---------------------------------------------------------------------------
__global__ __launch_bounds__(32)
void gemm_wmma_f16_kernel(const h16* __restrict__ A,
                          const h16* __restrict__ Bt,
                          const float* __restrict__ bias, int bias_n,
                          float* __restrict__ outF,
                          h16* __restrict__ outH,
                          int K, int Npad, int relu) {
    const int lane = threadIdx.x;
    const int grp  = lane >> 4;
    const int mn   = lane & 15;
    const int row0 = blockIdx.x * 32;
    const int col0 = blockIdx.y * 32;

    const h16* pa0 = A  + (size_t)(row0 + mn) * K + grp * 8;
    const h16* pa1 = pa0 + (size_t)16 * K;
    const h16* pb0 = Bt + (size_t)(col0 + mn) * K + grp * 8;
    const h16* pb1 = pb0 + (size_t)16 * K;

    v8f acc00 = {}, acc01 = {}, acc10 = {}, acc11 = {};

    v16h a0 = load_afrag(pa0);
    v16h a1 = load_afrag(pa1);
    v16h b0 = load_afrag(pb0);
    v16h b1 = load_afrag(pb1);

    for (int kb = 32; kb < K; kb += 32) {
        v16h a0n = load_afrag(pa0 + kb);
        v16h a1n = load_afrag(pa1 + kb);
        v16h b0n = load_afrag(pb0 + kb);
        v16h b1n = load_afrag(pb1 + kb);
        __builtin_prefetch(pa0 + kb + 32, 0, 1);   // speculative: safe past end
        __builtin_prefetch(pa1 + kb + 32, 0, 1);
        __builtin_prefetch(pb0 + kb + 32, 0, 1);
        __builtin_prefetch(pb1 + kb + 32, 0, 1);
        acc00 = wmma16(a0, b0, acc00);
        acc01 = wmma16(a0, b1, acc01);
        acc10 = wmma16(a1, b0, acc10);
        acc11 = wmma16(a1, b1, acc11);
        a0 = a0n; a1 = a1n; b0 = b0n; b1 = b1n;
    }
    acc00 = wmma16(a0, b0, acc00);
    acc01 = wmma16(a0, b1, acc01);
    acc10 = wmma16(a1, b0, acc10);
    acc11 = wmma16(a1, b1, acc11);

    const int n0 = col0 + mn;
    const int n1 = n0 + 16;
    float bv0 = (bias != nullptr && n0 < bias_n) ? bias[n0] : 0.0f;
    float bv1 = (bias != nullptr && n1 < bias_n) ? bias[n1] : 0.0f;
    #pragma unroll
    for (int r = 0; r < 8; ++r) {
        int m0 = row0 + (grp << 3) + r;
        int m1 = m0 + 16;
        float v00 = acc00[r] + bv0, v01 = acc01[r] + bv1;
        float v10 = acc10[r] + bv0, v11 = acc11[r] + bv1;
        if (relu) {
            v00 = fmaxf(v00, 0.0f); v01 = fmaxf(v01, 0.0f);
            v10 = fmaxf(v10, 0.0f); v11 = fmaxf(v11, 0.0f);
        }
        size_t o00 = (size_t)m0 * Npad + n0, o01 = (size_t)m0 * Npad + n1;
        size_t o10 = (size_t)m1 * Npad + n0, o11 = (size_t)m1 * Npad + n1;
        if (outF) { outF[o00] = v00; outF[o01] = v01; outF[o10] = v10; outF[o11] = v11; }
        if (outH) { outH[o00] = (h16)v00; outH[o01] = (h16)v01;
                    outH[o10] = (h16)v10; outH[o11] = (h16)v11; }
    }
}

// ---------------------------------------------------------------------------
// Per-row head: softmax -> conf/label, select deltas, decode+clip box
// ---------------------------------------------------------------------------
__global__ void head_kernel(const float* __restrict__ logits,
                            const float* __restrict__ deltas,
                            const float* __restrict__ props,
                            float* __restrict__ boxes,
                            float* __restrict__ conf,
                            int* __restrict__ lab,
                            const int* __restrict__ img_h_p,
                            const int* __restrict__ img_w_p) {
    int r = blockIdx.x * blockDim.x + threadIdx.x;
    if (r >= NPROP) return;
    const float* lg = logits + (size_t)r * LOGPAD;
    float best = -1e30f; int bl = 0;
    for (int k = 0; k < KCLS; ++k) {
        float v = lg[k];
        if (v > best) { best = v; bl = k; }
    }
    float z = 0.0f;
    for (int k = 0; k < KCLS; ++k) z += expf(lg[k] - best);
    float cf = 1.0f / z;

    const float* dd = deltas + (size_t)r * DELPAD + bl * 4;
    float px1 = props[r*4+0], py1 = props[r*4+1];
    float px2 = props[r*4+2], py2 = props[r*4+3];
    float w  = fmaxf(px2 - px1, 1.0f);
    float h  = fmaxf(py2 - py1, 1.0f);
    float cx = px1 + 0.5f * w;
    float cy = py1 + 0.5f * h;
    float cx_ = cx + dd[0] * w;
    float cy_ = cy + dd[1] * h;
    float w_  = w * fminf(expf(dd[2]), 10000.0f);
    float h_  = h * fminf(expf(dd[3]), 10000.0f);
    float img_w = (float)(*img_w_p), img_h = (float)(*img_h_p);
    float bx1 = fminf(fmaxf(cx_ - 0.5f*w_, 0.0f), img_w - 1.0f);
    float by1 = fminf(fmaxf(cy_ - 0.5f*h_, 0.0f), img_h - 1.0f);
    float bx2 = fminf(fmaxf(cx_ + 0.5f*w_, 0.0f), img_w - 1.0f);
    float by2 = fminf(fmaxf(cy_ + 0.5f*h_, 0.0f), img_h - 1.0f);
    boxes[r*4+0] = bx1; boxes[r*4+1] = by1;
    boxes[r*4+2] = bx2; boxes[r*4+3] = by2;
    conf[r] = (bl > 0) ? cf : 0.0f;
    lab[r]  = bl;
}

// ---------------------------------------------------------------------------
// Stable descending rank sort (O(N^2), N=600). Matches jnp.argsort(-conf).
// ---------------------------------------------------------------------------
__global__ __launch_bounds__(640)
void sort_kernel(const float* __restrict__ conf,
                 const float* __restrict__ boxes,
                 const int* __restrict__ lab,
                 float* __restrict__ s_s, float* __restrict__ b_s,
                 int* __restrict__ l_s) {
    int i = threadIdx.x;
    if (i >= NPROP) return;
    float ci = conf[i];
    int rank = 0;
    for (int j = 0; j < NPROP; ++j) {
        float cj = conf[j];
        rank += (cj > ci) || (cj == ci && j < i);
    }
    s_s[rank] = ci;
    l_s[rank] = lab[i];
    b_s[rank*4+0] = boxes[i*4+0];
    b_s[rank*4+1] = boxes[i*4+1];
    b_s[rank*4+2] = boxes[i*4+2];
    b_s[rank*4+3] = boxes[i*4+3];
}

// ---------------------------------------------------------------------------
// Greedy NMS with class offset, sequential suppression identical to ref loop
// ---------------------------------------------------------------------------
__global__ __launch_bounds__(640)
void nms_kernel(const float* __restrict__ b_s, const int* __restrict__ l_s,
                int* __restrict__ keep_out) {
    __shared__ float bx[NPROP][4];
    __shared__ float area[NPROP];
    __shared__ int   keep[NPROP];
    int j = threadIdx.x;
    if (j < NPROP) {
        float off = (float)l_s[j] * 10000.0f;
        float x1 = b_s[j*4+0] + off, y1 = b_s[j*4+1] + off;
        float x2 = b_s[j*4+2] + off, y2 = b_s[j*4+3] + off;
        bx[j][0] = x1; bx[j][1] = y1; bx[j][2] = x2; bx[j][3] = y2;
        area[j] = fmaxf(x2 - x1, 0.0f) * fmaxf(y2 - y1, 0.0f);
        keep[j] = 1;
    }
    __syncthreads();
    for (int i = 0; i < NPROP - 1; ++i) {
        if (j < NPROP && j > i && keep[i] && keep[j]) {
            float lx = fmaxf(bx[i][0], bx[j][0]);
            float ly = fmaxf(bx[i][1], bx[j][1]);
            float rx = fminf(bx[i][2], bx[j][2]);
            float ry = fminf(bx[i][3], bx[j][3]);
            float iw = fmaxf(rx - lx, 0.0f);
            float ih = fmaxf(ry - ly, 0.0f);
            float inter = iw * ih;
            float iou = inter / fmaxf(area[i] + area[j] - inter, 1e-9f);
            if (iou > 0.5f) keep[j] = 0;
        }
        __syncthreads();
    }
    if (j < NPROP) keep_out[j] = keep[j];
}

// ---------------------------------------------------------------------------
// Final compaction (equivalent to top_k on where(valid, s, -1) since s_s is
// already sorted descending): p-th valid sorted entry -> output slot p.
// ---------------------------------------------------------------------------
__global__ __launch_bounds__(1024)
void final_kernel(const float* __restrict__ b_s, const float* __restrict__ s_s,
                  const int* __restrict__ l_s, const int* __restrict__ keep,
                  float* __restrict__ out) {
    int t = threadIdx.x;
    for (int e = t; e < TOPK_N * 6; e += blockDim.x) out[e] = 0.0f; // zero bits
    __syncthreads();
    if (t < NPROP) {
        bool valid = keep[t] && (s_s[t] >= 0.05f) && (l_s[t] > 0);
        if (valid) {
            int pos = 0;
            for (int j = 0; j < t; ++j)
                pos += (keep[j] && (s_s[j] >= 0.05f) && (l_s[j] > 0)) ? 1 : 0;
            if (pos < TOPK_N) {
                out[pos*4+0] = b_s[t*4+0];
                out[pos*4+1] = b_s[t*4+1];
                out[pos*4+2] = b_s[t*4+2];
                out[pos*4+3] = b_s[t*4+3];
                out[TOPK_N*4 + pos] = s_s[t];
                ((int*)out)[TOPK_N*5 + pos] = l_s[t];
            }
        }
    }
}

// ---------------------------------------------------------------------------
extern "C" void kernel_launch(void* const* d_in, const int* in_sizes, int n_in,
                              void* d_out, int out_size, void* d_ws, size_t ws_size,
                              hipStream_t stream) {
    const float* fmap  = (const float*)d_in[0];
    const float* props = (const float*)d_in[1];
    const float* w1    = (const float*)d_in[2];
    const float* b1    = (const float*)d_in[3];
    const float* w2    = (const float*)d_in[4];
    const float* b2    = (const float*)d_in[5];
    const float* wc    = (const float*)d_in[6];
    const float* bc    = (const float*)d_in[7];
    const float* wb    = (const float*)d_in[8];
    const float* bb    = (const float*)d_in[9];
    const int*   img_h = (const int*)d_in[10];
    const int*   img_w = (const int*)d_in[11];

    char* ws = (char*)d_ws;
    size_t off = 0;
    auto alloc = [&](size_t bytes) -> void* {
        void* p = ws + off;
        off += (bytes + 255) & ~(size_t)255;
        return p;
    };

    h16*   A     = (h16*)  alloc((size_t)MPAD * DIN * sizeof(h16));
    h16*   W1t   = (h16*)  alloc((size_t)HID  * DIN * sizeof(h16));
    h16*   W2t   = (h16*)  alloc((size_t)HID  * HID * sizeof(h16));
    h16*   Wct   = (h16*)  alloc((size_t)LOGPAD * HID * sizeof(h16));
    h16*   Wbt   = (h16*)  alloc((size_t)DELPAD * HID * sizeof(h16));
    h16*   H1    = (h16*)  alloc((size_t)MPAD * HID * sizeof(h16));
    h16*   H2    = (h16*)  alloc((size_t)MPAD * HID * sizeof(h16));
    float* LOG   = (float*)alloc((size_t)MPAD * LOGPAD * sizeof(float));
    float* DEL   = (float*)alloc((size_t)MPAD * DELPAD * sizeof(float));
    float* boxes = (float*)alloc((size_t)NPROP * 4 * sizeof(float));
    float* conf  = (float*)alloc((size_t)NPROP * sizeof(float));
    int*   lab   = (int*)  alloc((size_t)NPROP * sizeof(int));
    float* b_s   = (float*)alloc((size_t)NPROP * 4 * sizeof(float));
    float* s_s   = (float*)alloc((size_t)NPROP * sizeof(float));
    int*   l_s   = (int*)  alloc((size_t)NPROP * sizeof(int));
    int*   keep  = (int*)  alloc((size_t)NPROP * sizeof(int));

    const int TB = 256;

    // Weight convert+transpose to fp16 (N_pad x K), LDS-tiled
    transpose_f16_kernel<<<dim3(DIN / 32, HID / 32), dim3(32, 8), 0, stream>>>(
        w1, W1t, DIN, HID, HID);
    transpose_f16_kernel<<<dim3(HID / 32, HID / 32), dim3(32, 8), 0, stream>>>(
        w2, W2t, HID, HID, HID);
    transpose_f16_kernel<<<dim3(HID / 32, LOGPAD / 32), dim3(32, 8), 0, stream>>>(
        wc, Wct, HID, KCLS, LOGPAD);
    transpose_f16_kernel<<<dim3(HID / 32, DELPAD / 32), dim3(32, 8), 0, stream>>>(
        wb, Wbt, HID, DELN, DELPAD);

    // Zero pad rows 600..607 of A
    {
        long cnt = (long)(MPAD - NPROP) * DIN;
        zero_h16_kernel<<<(cnt + TB - 1) / TB, TB, 0, stream>>>(A + (size_t)NPROP * DIN, cnt);
    }

    // ROI align -> fp16 A
    {
        long tot = (long)NPROP * CCH * 49;
        roi_align_kernel<<<(tot + TB - 1) / TB, TB, 0, stream>>>(fmap, props, A, img_h, img_w);
    }

    // FC1: (608 x 25088) x (25088 x 1024) + b1, relu -> H1 (f16)
    gemm_wmma_f16_kernel<<<dim3(MPAD / 32, HID / 32), 32, 0, stream>>>(
        A, W1t, b1, HID, nullptr, H1, DIN, HID, 1);
    // FC2: (608 x 1024) x (1024 x 1024) + b2, relu -> H2 (f16)
    gemm_wmma_f16_kernel<<<dim3(MPAD / 32, HID / 32), 32, 0, stream>>>(
        H1, W2t, b2, HID, nullptr, H2, HID, HID, 1);
    // Logits: (608 x 1024) x (1024 x 32pad) + bc -> LOG (f32)
    gemm_wmma_f16_kernel<<<dim3(MPAD / 32, LOGPAD / 32), 32, 0, stream>>>(
        H2, Wct, bc, KCLS, LOG, nullptr, HID, LOGPAD, 0);
    // Deltas: (608 x 1024) x (1024 x 96pad) + bb -> DEL (f32)
    gemm_wmma_f16_kernel<<<dim3(MPAD / 32, DELPAD / 32), 32, 0, stream>>>(
        H2, Wbt, bb, DELN, DEL, nullptr, HID, DELPAD, 0);

    // Softmax / argmax / delta decode / clip
    head_kernel<<<(NPROP + TB - 1) / TB, TB, 0, stream>>>(
        LOG, DEL, props, boxes, conf, lab, img_h, img_w);

    // Stable descending sort by confidence
    sort_kernel<<<1, 640, 0, stream>>>(conf, boxes, lab, s_s, b_s, l_s);

    // Class-offset greedy NMS
    nms_kernel<<<1, 640, 0, stream>>>(b_s, l_s, keep);

    // Final top-300 compaction into d_out (boxes | scores | int32 labels)
    final_kernel<<<1, 1024, 0, stream>>>(b_s, s_s, l_s, keep, (float*)d_out);
}
